// LMUFFT_24567212933841
// MI455X (gfx1250) — matrix operand order repaired and checked by reading, not compiled, not run on hardware
//
#include <hip/hip_runtime.h>

// ---------------------------------------------------------------------------
// LMU-FFT forward pass for MI455X (gfx1250, wave32, WMMA, TDM + async copy).
// Shapes: B=8, S=4096, I=512, M=512, H=1024, 2S=8192 (FFT length).
// ---------------------------------------------------------------------------

typedef __attribute__((ext_vector_type(16))) _Float16 v16h;
typedef __attribute__((ext_vector_type(8)))  float    v8f;
typedef __attribute__((ext_vector_type(4)))  _Float16 v4h;
typedef __attribute__((ext_vector_type(4)))  unsigned u32x4;
typedef __attribute__((ext_vector_type(8)))  unsigned u32x8;

union HFrag { uint4 u[2]; v16h v; };

#define Bsz   8
#define Slen  4096
#define Ilen  512
#define Mlen  512
#define Hlen  1024
#define N2    8192        // FFT length = 2S
#define NF    4097        // S+1 rfft bins
#define Kdim  1024        // M + I

// CDNA5 async global->LDS copy (ASYNCcnt-tracked, no VGPR round trip).
__device__ __forceinline__ void async_b128(unsigned lds_off, const void* g) {
  asm volatile("global_load_async_to_lds_b128 %0, %1, off"
               :: "v"(lds_off), "v"(g) : "memory");
}
__device__ __forceinline__ void wait_async0() {
  asm volatile("s_wait_asynccnt 0x0" ::: "memory");
}

// CDNA5 Tensor Data Mover: one 2-D tile DMA (global -> LDS), TENSORcnt-tracked.
// D# group0 (4 SGPRs) + group1 (8 SGPRs); VADDR2/3 omitted (<=2D tensor).
// tile: tile_d0 x tile_d1 elements of 2 bytes, row stride = stride_d0 elems.
__device__ __forceinline__ void tdm_load_2d_f16(unsigned lds_off,
                                                const void* gaddr,
                                                unsigned tensor_d0,
                                                unsigned tensor_d1,
                                                unsigned stride_d0,
                                                unsigned tile_d0,
                                                unsigned tile_d1) {
  unsigned long long ga = (unsigned long long)gaddr;
  u32x4 g0;
  g0[0] = 1u;                                            // count=1, user mode
  g0[1] = lds_off;                                       // lds_addr (bytes)
  g0[2] = (unsigned)(ga & 0xFFFFFFFFull);                // global_addr[31:0]
  g0[3] = (unsigned)((ga >> 32) & 0x01FFFFFFull)         // global_addr[56:32]
          | (2u << 30);                                  // type = 2 ("image")
  u32x8 g1;
  g1[0] = (1u << 16);                                    // data_size=1 -> 2B
  g1[1] = (tensor_d0 & 0xFFFFu) << 16;                   // dim0[15:0]
  g1[2] = (tensor_d0 >> 16) | ((tensor_d1 & 0xFFFFu) << 16);
  g1[3] = (tensor_d1 >> 16) | ((tile_d0 & 0xFFFFu) << 16);
  g1[4] = tile_d1 & 0xFFFFu;                             // tile_dim1, tile_dim2=0
  g1[5] = stride_d0;                                     // dim0_stride[31:0]
  g1[6] = 0u;                                            // stride hi, dim1_stride lo
  g1[7] = 0u;
  asm volatile("tensor_load_to_lds %0, %1" :: "s"(g0), "s"(g1) : "memory");
}

// ------------------------------ u = relu(x . Wu) ---------------------------
__global__ __launch_bounds__(256) void u_kernel(const float* __restrict__ x,
                                                const float* __restrict__ Wu,
                                                const float* __restrict__ Wub,
                                                float* __restrict__ u) {
  int tid  = threadIdx.x;
  int wave = tid >> 5, lane = tid & 31;
  int row  = blockIdx.x * 8 + wave;            // (b*S + s)
  const float* xr = x + (size_t)row * Ilen;
  float s = 0.f;
#pragma unroll
  for (int it = 0; it < 4; ++it) {
    int i = lane * 4 + it * 128;
    float4 v = *(const float4*)&xr[i];
    float4 w = *(const float4*)&Wu[i];
    s = fmaf(v.x, w.x, s); s = fmaf(v.y, w.y, s);
    s = fmaf(v.z, w.z, s); s = fmaf(v.w, w.w, s);
  }
#pragma unroll
  for (int off = 16; off > 0; off >>= 1) s += __shfl_xor(s, off, 32);
  if (lane == 0) u[row] = fmaxf(s + Wub[0], 0.f);
}

// --------------------- 8192-pt Stockham radix-2 FFT in LDS -----------------
__device__ __forceinline__ void fft8192(float* sre, float* sim,
                                        float* dre, float* dim,
                                        float sign, int tid,
                                        float** fre, float** fim) {
  int l = 4096;
#pragma unroll 1
  for (int st = 0; st < 13; ++st) {
    int m = 1 << st;
    for (int i = tid; i < 4096; i += 256) {
      int j = i >> st;
      int k = i & (m - 1);
      float ang = sign * 3.14159265358979f * (float)j / (float)l;
      float wsn, wcs;
      __sincosf(ang, &wsn, &wcs);
      int i0 = j * m + k;
      float ar = sre[i0],        ai = sim[i0];
      float br = sre[i0 + 4096], bi = sim[i0 + 4096];
      float drr = ar - br, dii = ai - bi;
      int o0 = 2 * j * m + k;
      dre[o0]     = ar + br;
      dim[o0]     = ai + bi;
      dre[o0 + m] = wcs * drr - wsn * dii;
      dim[o0 + m] = wcs * dii + wsn * drr;
    }
    __syncthreads();
    float* t;
    t = sre; sre = dre; dre = t;
    t = sim; sim = dim; dim = t;
    l >>= 1;
  }
  *fre = sre; *fim = sim;
}

// ------------------------- forward rFFT of u (per batch) -------------------
__global__ __launch_bounds__(256) void fft_fwd_kernel(const float* __restrict__ u,
                                                      float2* __restrict__ fu) {
  __shared__ float reA[N2], imA[N2], reB[N2], imB[N2];   // 128 KB of 320 KB LDS
  int tid = threadIdx.x;
  int b   = blockIdx.x;
  for (int f = tid; f < N2; f += 256) {
    reA[f] = (f < Slen) ? u[(size_t)b * Slen + f] : 0.f;
    imA[f] = 0.f;
  }
  __syncthreads();
  float *fre, *fim;
  fft8192(reA, imA, reB, imB, -1.f, tid, &fre, &fim);
  for (int f = tid; f < NF; f += 256)
    fu[(size_t)b * NF + f] = make_float2(fre[f], fim[f]);
}

// ------------- per (b, m-channel): fu * fH, inverse FFT, emit m ------------
__device__ __forceinline__ float2 cmul(float2 a, float2 b) {
  return make_float2(a.x * b.x - a.y * b.y, a.x * b.y + a.y * b.x);
}

__global__ __launch_bounds__(256) void conv_kernel(const float2* __restrict__ fu,
                                                   const float2* __restrict__ fH,
                                                   _Float16* __restrict__ mbuf) {
  __shared__ float reA[N2], imA[N2], reB[N2], imB[N2];
  int tid = threadIdx.x;
  int b   = blockIdx.x >> 9;       // / 512
  int mc  = blockIdx.x & 511;
  const float2* fub = fu + (size_t)b  * NF;
  const float2* fhr = fH + (size_t)mc * NF;
  for (int f = tid; f < N2; f += 256) {
    float2 t;
    if (f <= Slen) {
      t = cmul(fub[f], fhr[f]);
    } else {
      float2 c = cmul(fub[N2 - f], fhr[N2 - f]);
      t = make_float2(c.x, -c.y);                  // conjugate mirror
    }
    reA[f] = t.x; imA[f] = t.y;
  }
  __syncthreads();
  float *fre, *fim;
  fft8192(reA, imA, reB, imB, +1.f, tid, &fre, &fim);
  const float scale = 1.f / (float)N2;
  _Float16* mrow = mbuf + ((size_t)(b * Mlen + mc)) * Slen;
  for (int t = tid; t < Slen; t += 256)
    mrow[t] = (_Float16)(fre[t] * scale);
}

// ------------- transpose m (b,M,S) -> h_in[:, :M] (b*S, Kdim) f16 ----------
__global__ void mtrans_kernel(const _Float16* __restrict__ mbuf,
                              _Float16* __restrict__ hin) {
  __shared__ _Float16 tile[32][33];
  int tx = threadIdx.x, ty = threadIdx.y;
  int s0 = blockIdx.x * 32, m0 = blockIdx.y * 32, b = blockIdx.z;
#pragma unroll
  for (int q = 0; q < 4; ++q) {
    int mr = m0 + ty + q * 8;
    tile[ty + q * 8][tx] = mbuf[((size_t)(b * Mlen + mr)) * Slen + s0 + tx];
  }
  __syncthreads();
#pragma unroll
  for (int q = 0; q < 4; ++q) {
    int sr = s0 + ty + q * 8;
    hin[((size_t)(b * Slen + sr)) * Kdim + m0 + tx] = tile[tx][ty + q * 8];
  }
}

// ---------------- pack x (f32) into h_in[:, M:] as f16 ---------------------
__global__ void xpack_kernel(const float* __restrict__ x,
                             _Float16* __restrict__ hin) {
  size_t g = ((size_t)blockIdx.x * 256 + threadIdx.x) * 4;
  float4 v  = *(const float4*)&x[g];
  size_t bs = g >> 9;              // / I
  int    i  = (int)(g & 511);
  v4h h = {(_Float16)v.x, (_Float16)v.y, (_Float16)v.z, (_Float16)v.w};
  *(v4h*)&hin[bs * Kdim + Mlen + i] = h;
}

// ---------------- convert W_h (H x Kdim, f32) to f16 -----------------------
__global__ void wcvt_kernel(const float* __restrict__ w,
                            _Float16* __restrict__ wh) {
  size_t g = ((size_t)blockIdx.x * 256 + threadIdx.x) * 4;
  float4 v = *(const float4*)&w[g];
  v4h h = {(_Float16)v.x, (_Float16)v.y, (_Float16)v.z, (_Float16)v.w};
  *(v4h*)&wh[g] = h;
}

// -------- out = relu(h_in @ W_h^T + b): WMMA f32_16x16x32_f16 GEMM ---------
// Block tile 128x128, 8 waves in 2x4 grid, wave tile 64x32 = 4x2 WMMA tiles.
// K staged in 64-wide slabs, double-buffered LDS:
//   A slab: one TDM tensor_load_to_lds issued by wave 0 (TENSORcnt),
//   B slab: per-lane global_load_async_to_lds_b128   (ASYNCcnt),
// one barrier per slab.
__global__ __launch_bounds__(256) void gemm_kernel(const _Float16* __restrict__ hin,
                                                   const _Float16* __restrict__ whf,
                                                   const float* __restrict__ bias,
                                                   float* __restrict__ out) {
  __shared__ __align__(16) _Float16 As[2][128 * 64];   // 2 x 16 KB
  __shared__ __align__(16) _Float16 Bs[2][128 * 64];   // 2 x 16 KB
  int tid  = threadIdx.x;
  int lane = tid & 31, wave = tid >> 5;
  int wm = wave & 1, wn = wave >> 1;
  int row0 = blockIdx.x * 128, col0 = blockIdx.y * 128;
  int wrow0 = wm * 64, wcol0 = wn * 32;
  int lrow = lane & 15, lhalf = lane >> 4;

  v8f acc[4][2] = {};

  // A slab: single TDM descriptor (128 rows x 64 halfs, row stride Kdim).
  auto stage_a = [&](int buf, int k0) {
    if (wave == 0) {
      tdm_load_2d_f16((unsigned)(unsigned long long)(const void*)&As[buf][0],
                      &hin[(size_t)row0 * Kdim + k0],
                      /*tensor_d0=*/Kdim, /*tensor_d1=*/Bsz * Slen,
                      /*stride_d0=*/Kdim,
                      /*tile_d0=*/64, /*tile_d1=*/128);
    }
  };
  // B slab: per-lane async b128 copies.
  auto stage_b = [&](int buf, int k0) {
#pragma unroll
    for (int c = 0; c < 4; ++c) {
      int e  = (tid + c * 256) * 8;     // 8 halfs per b128
      int r  = e >> 6;                  // row within 128
      int kk = e & 63;                  // k within slab
      async_b128((unsigned)(unsigned long long)(const void*)&Bs[buf][e],
                 &whf[(size_t)(col0 + r) * Kdim + k0 + kk]);
    }
  };
  auto wait_stage = [&]() {
    if (wave == 0) __builtin_amdgcn_s_wait_tensorcnt(0);  // TDM landed
    wait_async0();                                        // async B landed
    __syncthreads();
  };

  stage_a(0, 0);
  stage_b(0, 0);
  wait_stage();

  int buf = 0;
  for (int k0 = 0; k0 < Kdim; k0 += 64) {
    if (k0 + 64 < Kdim) {
      stage_a(buf ^ 1, k0 + 64);
      stage_b(buf ^ 1, k0 + 64);
    }

    // Warm L2 two slabs ahead (clamped, branchless).
    int kp = k0 + 128; if (kp > Kdim - 64) kp = Kdim - 64;
    __builtin_prefetch(&hin[(size_t)(row0 + (tid >> 2)) * Kdim + kp], 0, 1);
    __builtin_prefetch(&whf[(size_t)(col0 + (tid >> 2)) * Kdim + kp], 0, 1);

#pragma unroll
    for (int ks = 0; ks < 64; ks += 32) {
      // A fragments: 16x32 f16; lanes 0-15: M=0..15, K {0..7,16..23};
      // lanes 16-31: K {8..15,24..31} (ISA 7.12.2 16-bit A layout).
      HFrag a[4];
#pragma unroll
      for (int tm = 0; tm < 4; ++tm) {
        int ar = wrow0 + tm * 16 + lrow;
        int ka = ks + 8 * lhalf;
        a[tm].u[0] = *(const uint4*)&As[buf][ar * 64 + ka];
        a[tm].u[1] = *(const uint4*)&As[buf][ar * 64 + ka + 16];
      }
      // B fragments: 32x16 f16; lane -> N=lane%16, K {0..7,16..23} + 8*(lane/16).
      HFrag bf[2];
#pragma unroll
      for (int tn = 0; tn < 2; ++tn) {
        int bc = wcol0 + tn * 16 + lrow;
        int kb = ks + 8 * lhalf;
        bf[tn].u[0] = *(const uint4*)&Bs[buf][bc * 64 + kb];
        bf[tn].u[1] = *(const uint4*)&Bs[buf][bc * 64 + kb + 16];
      }
#pragma unroll
      for (int tm = 0; tm < 4; ++tm)
#pragma unroll
        for (int tn = 0; tn < 2; ++tn)
          acc[tm][tn] = __builtin_amdgcn_wmma_f32_16x16x32_f16(
              false, a[tm].v, false, bf[tn].v, (short)0, acc[tm][tn],
              false, false);
    }

    wait_stage();        // next slab fully landed, everyone done with current
    buf ^= 1;
  }

  // Epilogue: C layout (ISA 7.12.2): VGPR r -> M = r + 8*(lane/16), N = lane%16.
#pragma unroll
  for (int tm = 0; tm < 4; ++tm)
#pragma unroll
    for (int tn = 0; tn < 2; ++tn)
#pragma unroll
      for (int r = 0; r < 8; ++r) {
        int Ml  = r + lhalf * 8;
        int row = row0 + wrow0 + tm * 16 + Ml;
        int col = col0 + wcol0 + tn * 16 + lrow;
        out[(size_t)row * Hlen + col] = fmaxf(acc[tm][tn][r] + bias[col], 0.f);
      }
}

// ---------------------------------------------------------------------------
extern "C" void kernel_launch(void* const* d_in, const int* in_sizes, int n_in,
                              void* d_out, int out_size, void* d_ws, size_t ws_size,
                              hipStream_t stream) {
  const float*  x   = (const float*)d_in[0];
  const float*  Wu  = (const float*)d_in[1];
  const float*  Wub = (const float*)d_in[2];
  const float*  Wh  = (const float*)d_in[3];
  const float*  Whb = (const float*)d_in[4];
  const float2* fH  = (const float2*)d_in[5];
  float* out = (float*)d_out;

  // Workspace layout (256-B aligned slabs).
  char* ws = (char*)d_ws;
  size_t off = 0;
  auto alloc = [&](size_t bytes) { size_t o = off; off = (off + bytes + 255) & ~(size_t)255; return o; };
  size_t OFF_U   = alloc((size_t)Bsz * Slen * sizeof(float));          // u
  size_t OFF_FU  = alloc((size_t)Bsz * NF * sizeof(float2));           // rfft(u)
  size_t OFF_M   = alloc((size_t)Bsz * Mlen * Slen * sizeof(_Float16));// m (b,M,S)
  size_t OFF_HIN = alloc((size_t)Bsz * Slen * Kdim * sizeof(_Float16));// [m|x] f16
  size_t OFF_W   = alloc((size_t)Hlen * Kdim * sizeof(_Float16));      // W_h f16

  float*    u    = (float*)(ws + OFF_U);
  float2*   fu   = (float2*)(ws + OFF_FU);
  _Float16* mbuf = (_Float16*)(ws + OFF_M);
  _Float16* hin  = (_Float16*)(ws + OFF_HIN);
  _Float16* whf  = (_Float16*)(ws + OFF_W);

  // 1) u = relu(x . Wu + b)                : 32768 rows, 1 wave32 each
  u_kernel<<<dim3((Bsz * Slen) / 8), dim3(256), 0, stream>>>(x, Wu, Wub, u);

  // 2) rfft(u) per batch                   : 8 blocks, 128 KB LDS each
  fft_fwd_kernel<<<dim3(Bsz), dim3(256), 0, stream>>>(u, fu);

  // 3) per (b, m): spectrum mult + irfft   : 4096 blocks, 128 KB LDS each
  conv_kernel<<<dim3(Bsz * Mlen), dim3(256), 0, stream>>>(fu, fH, mbuf);

  // 4) transpose m into h_in[:, :M] (f16)
  mtrans_kernel<<<dim3(Slen / 32, Mlen / 32, Bsz), dim3(32, 8), 0, stream>>>(mbuf, hin);

  // 5) pack x into h_in[:, M:] (f16)
  xpack_kernel<<<dim3((Bsz * Slen * Ilen) / (256 * 4)), dim3(256), 0, stream>>>(x, hin);

  // 6) W_h -> f16
  wcvt_kernel<<<dim3((Hlen * Kdim) / (256 * 4)), dim3(256), 0, stream>>>(Wh, whf);

  // 7) out = relu(h_in @ W_h^T + b) via WMMA, 256x8 blocks of 128x128
  gemm_kernel<<<dim3((Bsz * Slen) / 128, Hlen / 128), dim3(256), 0, stream>>>(hin, whf, Whb, out);

  (void)in_sizes; (void)n_in; (void)out_size; (void)ws_size;
}
